// PointerNet_68118181314679
// MI455X (gfx1250) — matrix operand order repaired and checked by reading, not compiled, not run on hardware
//
#include <hip/hip_runtime.h>

#define BATCH   65536
#define NOBJ    20
#define NDIM    16
#define HH      64
#define WAVES   2
#define ROWS    32        // padded M (2 x 16-row tiles)
#define LDH     68        // padded leading dim: 68*4B = 272B, 16B aligned rows
#define NEGINF  -1.0e9f

// prepped weight fragment table in d_ws (f16), 512 halves (1KB) per fragment
#define FRAG_HALVES 512
#define FE   0    // w_embed : 4 frags  (nkt=1, K padded 16->32)
#define F1A  4    // w_mp1a  : 16 frags (nkt=4)
#define F1B  20   // w_mp1b  : 8 frags  (nkt=2)
#define F2A  28   // w_mp2a  : 16 frags (nkt=4)
#define F2B  44   // w_mp2b  : 8 frags  (nkt=2)
#define FK   52   // w_k     : 8 frags  (nkt=2)
#define NFRAGS 60

typedef __attribute__((ext_vector_type(16))) _Float16 v16h;
typedef __attribute__((ext_vector_type(8)))  float    v8f;

// ---------------- weight prep: f32 [K][64] -> f16 B-fragment layout ----------------
// B 32x16 f16 layout: lane = column (lane&15), grp = lane>>4;
// half i of lane's v16h = K = kt*32 + (i<8 ? grp*8+i : 16+grp*8+(i-8)). One fragment = 32 lanes * 32B = 1KB.
__global__ void prep_weights(const float* __restrict__ w_embed,
                             const float* __restrict__ w_mp1a,
                             const float* __restrict__ w_mp1b,
                             const float* __restrict__ w_mp2a,
                             const float* __restrict__ w_mp2b,
                             const float* __restrict__ w_k,
                             _Float16* __restrict__ dst) {
  int f    = blockIdx.x;      // 0..59
  int lane = threadIdx.x;     // 0..31
  const float* W; int Ksrc; int fl;
  if      (f < F1A) { W = w_embed; Ksrc = 16;  fl = f;        }
  else if (f < F1B) { W = w_mp1a;  Ksrc = 128; fl = f - F1A;  }
  else if (f < F2A) { W = w_mp1b;  Ksrc = 64;  fl = f - F1B;  }
  else if (f < F2B) { W = w_mp2a;  Ksrc = 128; fl = f - F2A;  }
  else if (f < FK)  { W = w_mp2b;  Ksrc = 64;  fl = f - F2B;  }
  else              { W = w_k;     Ksrc = 64;  fl = f - FK;   }
  int kt = fl >> 2, nt = fl & 3;
  int n   = nt * 16 + (lane & 15);
  int grp = lane >> 4;
  _Float16* o = dst + (size_t)f * FRAG_HALVES + lane * 16;
#pragma unroll
  for (int i = 0; i < 16; ++i) {
    int k = kt * 32 + ((i < 8) ? (grp * 8 + i) : (16 + grp * 8 + (i - 8)));
    o[i] = (_Float16)((k < Ksrc) ? W[k * HH + n] : 0.0f);
  }
}

// ---------------- fragment builders (all unconditional) ----------------

// 16-bit A 16x32: lane reads two contiguous 8-float runs (k = kbase+grp*8+{0..7}, +16)
__device__ __forceinline__ v16h a_frag(const float* A, int lda, int rowbase, int kbase, int lane) {
  const int m   = rowbase + (lane & 15);
  const int grp = lane >> 4;
  const float* p = A + m * lda + kbase + grp * 8;
  float4 x0 = *(const float4*)(p);
  float4 x1 = *(const float4*)(p + 4);
  float4 y0 = *(const float4*)(p + 16);
  float4 y1 = *(const float4*)(p + 20);
  v16h a;
  a[0]  = (_Float16)x0.x; a[1]  = (_Float16)x0.y; a[2]  = (_Float16)x0.z; a[3]  = (_Float16)x0.w;
  a[4]  = (_Float16)x1.x; a[5]  = (_Float16)x1.y; a[6]  = (_Float16)x1.z; a[7]  = (_Float16)x1.w;
  a[8]  = (_Float16)y0.x; a[9]  = (_Float16)y0.y; a[10] = (_Float16)y0.z; a[11] = (_Float16)y0.w;
  a[12] = (_Float16)y1.x; a[13] = (_Float16)y1.y; a[14] = (_Float16)y1.z; a[15] = (_Float16)y1.w;
  return a;
}

// B fragment: 32 contiguous bytes per lane from prepped global table
__device__ __forceinline__ v16h b_frag(const _Float16* wf, int kt, int nt, int lane) {
  return *(const v16h*)(wf + (size_t)(kt * 4 + nt) * FRAG_HALVES + lane * 16);
}

// C/D 16x16 f32: VGPR r -> M = (lane>=16?8:0)+r, N = lane&15. Unconditional stores:
// padding rows (20..31) only ever affect their own discarded rows.
template <bool ACC>
__device__ __forceinline__ void d_store(float* D, int ldd, int rowbase, int ncol, int lane, v8f c) {
  const int m0 = rowbase + ((lane >> 4) << 3);
  const int n  = ncol + (lane & 15);
#pragma unroll
  for (int r = 0; r < 8; ++r) {
    float* p = D + (m0 + r) * ldd + n;
    if (ACC) *p += c[r]; else *p = c[r];
  }
}

// D[32x64] = A[32xK] @ W[Kx64] + bias (W from prepped frag table)
template <bool ACC>
__device__ __forceinline__ void gemm32(const float* A, int lda, int nkt,
                                       const _Float16* wf, const float* bias,
                                       float* D, int ldd, int lane) {
  for (int mt = 0; mt < 2; ++mt) {
    for (int nt = 0; nt < 4; ++nt) {
      float bv = bias[nt * 16 + (lane & 15)];
      v8f c = {bv, bv, bv, bv, bv, bv, bv, bv};
      for (int kt = 0; kt < nkt; ++kt) {
        v16h a = a_frag(A, lda, mt * 16, kt * 32, lane);
        v16h b = b_frag(wf, kt, nt, lane);
        c = __builtin_amdgcn_wmma_f32_16x16x32_f16(false, a, false, b, (short)0, c, false, false);
      }
      d_store<ACC>(D, ldd, mt * 16, nt * 16, lane, c);
    }
  }
}

// D[32x64] = relu( [H | broadcast(msg)] (32x128) @ W[128x64] + bias )
__device__ __forceinline__ void gemm_cat_relu(const float* Hb, int lda, const float* msg,
                                              const _Float16* wf, const float* bias,
                                              float* D, int ldd, int lane) {
  for (int mt = 0; mt < 2; ++mt) {
    for (int nt = 0; nt < 4; ++nt) {
      float bv = bias[nt * 16 + (lane & 15)];
      v8f c = {bv, bv, bv, bv, bv, bv, bv, bv};
      for (int kt = 0; kt < 4; ++kt) {
        v16h a = (kt < 2) ? a_frag(Hb, lda, mt * 16, kt * 32, lane)
                          : a_frag(msg, 0, 0, (kt - 2) * 32, lane);  // lda=0 -> row-broadcast
        v16h b = b_frag(wf, kt, nt, lane);
        c = __builtin_amdgcn_wmma_f32_16x16x32_f16(false, a, false, b, (short)0, c, false, false);
      }
#pragma unroll
      for (int r = 0; r < 8; ++r) c[r] = fmaxf(c[r], 0.0f);
      d_store<false>(D, ldd, mt * 16, nt * 16, lane, c);
    }
  }
}

__global__ void __launch_bounds__(32 * WAVES)
pointer_net_kernel(const float* __restrict__ nf, const int* __restrict__ nn_count,
                   const float* __restrict__ b_embed,
                   const float* __restrict__ b_mp1a, const float* __restrict__ b_mp1b,
                   const float* __restrict__ b_mp2a, const float* __restrict__ b_mp2b,
                   const float* __restrict__ g1, const float* __restrict__ be1,
                   const float* __restrict__ g2, const float* __restrict__ be2,
                   const float* __restrict__ w_pool, const float* __restrict__ b_pool,
                   const float* __restrict__ w_op, const float* __restrict__ b_op,
                   const float* __restrict__ w_c1, const float* __restrict__ b_c1,
                   const float* __restrict__ w_c2, const float* __restrict__ b_c2,
                   const float* __restrict__ w_q, const float* __restrict__ b_q,
                   const float* __restrict__ b_k,
                   const _Float16* __restrict__ wfrag,
                   float* __restrict__ out) {
  __shared__ __align__(16) float s_h[WAVES][ROWS * LDH];
  __shared__ __align__(16) float s_t[WAVES][ROWS * LDH];
  __shared__ __align__(16) float s_msg[WAVES][HH];
  __shared__ __align__(16) float s_pool[WAVES][HH];
  __shared__ __align__(16) float s_g[WAVES][HH];
  __shared__ __align__(16) float s_q[WAVES][HH];

  const int lane = threadIdx.x & 31;
  const int wid  = threadIdx.x >> 5;
  const int b    = blockIdx.x * WAVES + wid;

  float* hb = s_h[wid];
  float* tb = s_t[wid];
  float* mg = s_msg[wid];
  float* pv = s_pool[wid];
  float* gv = s_g[wid];
  float* qv = s_q[wid];

  int count = nn_count[b];
  count = count < 0 ? 0 : (count > NOBJ ? NOBJ : count);
  const float invden = 1.0f / (float)(count > 0 ? count : 1);

  // ---- stage nf [20x16] into tb rows 0..19, cols 16..31 zeroed (K pad safety) ----
  {
    const float4* nfb = (const float4*)(nf + (size_t)b * NOBJ * NDIM);
    for (int idx = lane; idx < NOBJ * 8; idx += 32) {       // 20 rows x 8 float4 (32 cols)
      int row = idx >> 3, seg = idx & 7;
      float4 v = {0.f, 0.f, 0.f, 0.f};
      if (seg < 4) v = nfb[row * 4 + seg];
      *(float4*)(tb + row * LDH + seg * 4) = v;
    }
  }
  __syncthreads();

  // ---- embed: h = nf @ w_embed + b_embed ----
  gemm32<false>(tb, LDH, 1, wfrag + (size_t)FE * FRAG_HALVES, b_embed, hb, LDH, lane);
  __syncthreads();

  // ============ message-pass block 1 ============
  for (int c = lane; c < HH; c += 32) {
    float s = 0.0f;
    for (int r = 0; r < count; ++r) s += hb[r * LDH + c];
    mg[c] = s * invden;
  }
  __syncthreads();
  gemm_cat_relu(hb, LDH, mg, wfrag + (size_t)F1A * FRAG_HALVES, b_mp1a, tb, LDH, lane);
  __syncthreads();
  gemm32<true>(tb, LDH, 2, wfrag + (size_t)F1B * FRAG_HALVES, b_mp1b, hb, LDH, lane);
  __syncthreads();
  if (lane < NOBJ) {
    float* row = &hb[lane * LDH];
    float mean = 0.0f;
    for (int c = 0; c < HH; ++c) mean += row[c];
    mean *= (1.0f / HH);
    float var = 0.0f;
    for (int c = 0; c < HH; ++c) { float d = row[c] - mean; var += d * d; }
    var *= (1.0f / HH);
    float inv = rsqrtf(var + 1e-5f);
    float mk = (lane < count) ? 1.0f : 0.0f;
    for (int c = 0; c < HH; ++c)
      row[c] = ((row[c] - mean) * inv * g1[c] + be1[c]) * mk;
  }
  __syncthreads();

  // ============ message-pass block 2 ============
  for (int c = lane; c < HH; c += 32) {
    float s = 0.0f;
    for (int r = 0; r < count; ++r) s += hb[r * LDH + c];
    mg[c] = s * invden;
  }
  __syncthreads();
  gemm_cat_relu(hb, LDH, mg, wfrag + (size_t)F2A * FRAG_HALVES, b_mp2a, tb, LDH, lane);
  __syncthreads();
  gemm32<true>(tb, LDH, 2, wfrag + (size_t)F2B * FRAG_HALVES, b_mp2b, hb, LDH, lane);
  __syncthreads();
  if (lane < NOBJ) {
    float* row = &hb[lane * LDH];
    float mean = 0.0f;
    for (int c = 0; c < HH; ++c) mean += row[c];
    mean *= (1.0f / HH);
    float var = 0.0f;
    for (int c = 0; c < HH; ++c) { float d = row[c] - mean; var += d * d; }
    var *= (1.0f / HH);
    float inv = rsqrtf(var + 1e-5f);
    float mk = (lane < count) ? 1.0f : 0.0f;
    for (int c = 0; c < HH; ++c)
      row[c] = ((row[c] - mean) * inv * g2[c] + be2[c]) * mk;
  }
  __syncthreads();

  // ---- k = h @ w_k + b_k  -> tb ----
  gemm32<false>(hb, LDH, 2, wfrag + (size_t)FK * FRAG_HALVES, b_k, tb, LDH, lane);

  // ---- pooled mean (masked rows of h are already zero) ----
  for (int c = lane; c < HH; c += 32) {
    float s = 0.0f;
    for (int r = 0; r < NOBJ; ++r) s += hb[r * LDH + c];
    pv[c] = s * invden;
  }
  __syncthreads();

  // ---- g = relu(pooled @ w_pool + b_pool) ----
  for (int c = lane; c < HH; c += 32) {
    float s = b_pool[c];
    for (int k = 0; k < HH; ++k) s += pv[k] * w_pool[k * HH + c];
    gv[c] = fmaxf(s, 0.0f);
  }
  __syncthreads();

  // ---- q = g @ w_q + b_q ----
  for (int c = lane; c < HH; c += 32) {
    float s = b_q[c];
    for (int k = 0; k < HH; ++k) s += gv[k] * w_q[k * HH + c];
    qv[c] = s;
  }

  // ---- heads ----
  if (lane < 8) {
    float s = b_op[lane];
    for (int k = 0; k < HH; ++k) s += gv[k] * w_op[k * 8 + lane];
    out[(size_t)b * 8 + lane] = s;
  }
  if (lane < 10) {
    float s1 = b_c1[lane], s2 = b_c2[lane];
    for (int k = 0; k < HH; ++k) {
      s1 += gv[k] * w_c1[k * 10 + lane];
      s2 += gv[k] * w_c2[k * 10 + lane];
    }
    out[(size_t)BATCH * 8  + (size_t)b * 10 + lane] = s1;
    out[(size_t)BATCH * 18 + (size_t)b * 10 + lane] = s2;
  }
  __syncthreads();

  // ---- pointer logits: ptr[n] = q . k_n, masked ----
  if (lane < NOBJ) {
    float s = 0.0f;
    for (int c = 0; c < HH; ++c) s += qv[c] * tb[lane * LDH + c];
    out[(size_t)BATCH * 28 + (size_t)b * NOBJ + lane] = (lane < count) ? s : NEGINF;
  }
}

extern "C" void kernel_launch(void* const* d_in, const int* in_sizes, int n_in,
                              void* d_out, int out_size, void* d_ws, size_t ws_size,
                              hipStream_t stream) {
  const float* nf       = (const float*)d_in[0];
  const int*   nn_count = (const int*)  d_in[1];
  const float* w_embed  = (const float*)d_in[2];
  const float* b_embed  = (const float*)d_in[3];
  const float* w_mp1a   = (const float*)d_in[4];
  const float* b_mp1a   = (const float*)d_in[5];
  const float* w_mp1b   = (const float*)d_in[6];
  const float* b_mp1b   = (const float*)d_in[7];
  const float* w_mp2a   = (const float*)d_in[8];
  const float* b_mp2a   = (const float*)d_in[9];
  const float* w_mp2b   = (const float*)d_in[10];
  const float* b_mp2b   = (const float*)d_in[11];
  const float* g1       = (const float*)d_in[12];
  const float* be1      = (const float*)d_in[13];
  const float* g2       = (const float*)d_in[14];
  const float* be2      = (const float*)d_in[15];
  const float* w_pool   = (const float*)d_in[16];
  const float* b_pool   = (const float*)d_in[17];
  const float* w_op     = (const float*)d_in[18];
  const float* b_op     = (const float*)d_in[19];
  const float* w_c1     = (const float*)d_in[20];
  const float* b_c1     = (const float*)d_in[21];
  const float* w_c2     = (const float*)d_in[22];
  const float* b_c2     = (const float*)d_in[23];
  const float* w_q      = (const float*)d_in[24];
  const float* b_q      = (const float*)d_in[25];
  const float* w_k      = (const float*)d_in[26];
  const float* b_k      = (const float*)d_in[27];

  _Float16* wfrag = (_Float16*)d_ws;   // 60 frags * 1KB = 60KB

  prep_weights<<<dim3(NFRAGS), dim3(32), 0, stream>>>(
      w_embed, w_mp1a, w_mp1b, w_mp2a, w_mp2b, w_k, wfrag);

  dim3 grid(BATCH / WAVES);
  dim3 block(32 * WAVES);
  pointer_net_kernel<<<grid, block, 0, stream>>>(
      nf, nn_count, b_embed, b_mp1a, b_mp1b, b_mp2a, b_mp2b,
      g1, be1, g2, be2, w_pool, b_pool, w_op, b_op,
      w_c1, b_c1, w_c2, b_c2, w_q, b_q, b_k, wfrag, (float*)d_out);
}
